// SMG_2h_84000970375420
// MI455X (gfx1250) — matrix operand order repaired and checked by reading, not compiled, not run on hardware
//
#include <hip/hip_runtime.h>
#include <hip/hip_bf16.h>

// ---------------------------------------------------------------------------
// MI455X (gfx1250, wave32) masked-GraphConv GNN.
// Edge/node GEMMs on v_wmma_f32_16x16x32_bf16. Weights are pre-converted to
// bf16 AND pre-swizzled (in global memory) into fragment-native order, then
// hoisted into registers once per kernel (loop-invariant B operands).
// Feature gathers staged with async global->LDS.
// ---------------------------------------------------------------------------

typedef __attribute__((ext_vector_type(16))) __bf16 v16bf;
typedef __attribute__((ext_vector_type(8)))  __bf16 v8bf;
typedef __attribute__((ext_vector_type(8)))  float  v8f;
typedef __attribute__((ext_vector_type(4)))  int    v4i;

#if defined(__AMDGCN__) && __has_builtin(__builtin_amdgcn_global_load_async_to_lds_b128)
#define USE_ASYNC_LDS 1
#endif

__device__ __forceinline__ void wait_async_lds() {
#ifdef USE_ASYNC_LDS
#if __has_builtin(__builtin_amdgcn_s_wait_asynccnt)
  __builtin_amdgcn_s_wait_asynccnt(0);
#else
  asm volatile("s_wait_asynccnt 0" ::: "memory");
#endif
#endif
}

__device__ __forceinline__ unsigned short f32_to_bf16_rte(float f) {
  unsigned int u = __float_as_uint(f);
  u += 0x7FFFu + ((u >> 16) & 1u);
  return (unsigned short)(u >> 16);
}

__device__ __forceinline__ v8f wmma_bf16(v16bf a, v16bf b, v8f c) {
  // D = A(16x32 bf16) * B(32x16 bf16) + C(16x16 f32)
  return __builtin_amdgcn_wmma_f32_16x16x32_bf16(
      /*neg_a=*/false, a, /*neg_b=*/false, b,
      /*c_mod=*/(short)0, c, /*reuse_a=*/false, /*reuse_b=*/false);
}

// A fragment: 16x32 bf16 from LDS row-major [16][ldk] at kbase.
// Lanes 0-15 row=lane hold K {0..7,16..23}; lanes 16-31 hold K {8..15,24..31}.
__device__ __forceinline__ v16bf load_a_frag(const unsigned short* sA, int ldk,
                                             int kbase, int lane) {
  const int row = lane & 15;
  const int kh  = (lane >> 4) ? 8 : 0;
  const unsigned short* p = sA + row * ldk + kbase + kh;
  v8bf lo = *(const v8bf*)(p);
  v8bf hi = *(const v8bf*)(p + 16);
  return __builtin_shufflevector(lo, hi, 0, 1, 2, 3, 4, 5, 6, 7,
                                 8, 9, 10, 11, 12, 13, 14, 15);
}

// Load all B fragments of a pre-swizzled [KD x 64] weight into registers.
// Swizzled global layout: frag(kb,nt) -> 32 lanes x 16 contiguous bf16.
template <int KD>
__device__ __forceinline__ void load_b_frags(v16bf* bf,
                                             const unsigned short* __restrict__ wsw,
                                             int lane) {
#pragma unroll
  for (int f = 0; f < (KD / 32) * 4; ++f)
    bf[f] = *(const v16bf*)(wsw + (f << 9) + (lane << 4));
}

__device__ __forceinline__ void zero_acc4(v8f* acc) {
#pragma unroll
  for (int nt = 0; nt < 4; ++nt)
#pragma unroll
    for (int j = 0; j < 8; ++j) acc[nt][j] = 0.0f;
}

// Wave-level GEMM: [16 x KD] (LDS bf16) x [KD x 64] (register B) -> acc[4]
template <int KD>
__device__ __forceinline__ void gemm_16x64_reg(v8f* acc, const unsigned short* sArow,
                                               const v16bf* bf, int lane) {
#pragma unroll
  for (int kb = 0; kb < KD / 32; ++kb) {
    v16bf a = load_a_frag(sArow, KD, kb * 32, lane);
#pragma unroll
    for (int nt = 0; nt < 4; ++nt)
      acc[nt] = wmma_bf16(a, bf[kb * 4 + nt], acc[nt]);
  }
}

// Copy 16B chunks global->LDS (async where available)
__device__ __forceinline__ void copy16_g2l(unsigned short* lp,
                                           const unsigned short* gp) {
#ifdef USE_ASYNC_LDS
  __builtin_amdgcn_global_load_async_to_lds_b128(
      (__attribute__((address_space(1))) v4i*)gp,
      (__attribute__((address_space(3))) v4i*)lp, 0, 0);
#else
  *(uint4*)lp = *(const uint4*)gp;
#endif
}

// Stage 16 edges: rows r=0..15, cols [0:64)=feat[dst[e]], [64:128)=feat[src[e]]
__device__ __forceinline__ void stage_edge_tile(
    unsigned short* sArow, int* sDstRow,
    const unsigned short* __restrict__ feat,
    const int* __restrict__ src, const int* __restrict__ dst,
    int tile, int E_, int lane) {
  const int r = lane >> 1;       // two lanes per row
  const int half = lane & 1;     // 0 -> dst features, 1 -> src features
  const int e = tile * 16 + r;
  int node = -1;
  if (e < E_) node = half ? src[e] : dst[e];
  if (half == 0) sDstRow[r] = (e < E_) ? dst[e] : -1;
  unsigned short* dp = sArow + r * 128 + half * 64;
  if (node >= 0) {
    const unsigned short* gp = feat + node * 64;
#pragma unroll
    for (int j = 0; j < 8; ++j) copy16_g2l(dp + j * 8, gp + j * 8);
  } else {
    uint4 z = make_uint4(0u, 0u, 0u, 0u);
#pragma unroll
    for (int j = 0; j < 8; ++j) ((uint4*)dp)[j] = z;
  }
}

// Stage 16 nodes of f32 features [N][64] -> bf16 LDS [16][64]
__device__ __forceinline__ void stage_node_tile_f32(
    unsigned short* sArow, const float* __restrict__ feat, int tile, int N_, int lane) {
  const int r = lane >> 1;
  const int half = lane & 1;
  const int n = tile * 16 + r;
  unsigned short* dp = sArow + r * 64 + half * 32;
  if (n < N_) {
    const float4* sp = (const float4*)(feat + n * 64 + half * 32);
#pragma unroll
    for (int j = 0; j < 8; ++j) {
      float4 v = sp[j];
      dp[j * 4 + 0] = f32_to_bf16_rte(v.x);
      dp[j * 4 + 1] = f32_to_bf16_rte(v.y);
      dp[j * 4 + 2] = f32_to_bf16_rte(v.z);
      dp[j * 4 + 3] = f32_to_bf16_rte(v.w);
    }
  } else {
#pragma unroll
    for (int j = 0; j < 32; ++j) dp[j] = 0;
  }
}

// Stage 16 nodes of bf16 features [N][64] -> LDS [16][64]
__device__ __forceinline__ void stage_node_tile_bf16(
    unsigned short* sArow, const unsigned short* __restrict__ feat,
    int tile, int N_, int lane) {
  const int r = lane >> 1;
  const int half = lane & 1;
  const int n = tile * 16 + r;
  unsigned short* dp = sArow + r * 64 + half * 32;
  if (n < N_) {
    const unsigned short* gp = feat + n * 64 + half * 32;
#pragma unroll
    for (int j = 0; j < 4; ++j) copy16_g2l(dp + j * 8, gp + j * 8);
  } else {
    uint4 z = make_uint4(0u, 0u, 0u, 0u);
#pragma unroll
    for (int j = 0; j < 4; ++j) ((uint4*)dp)[j] = z;
  }
}

// ---------------------------------------------------------------------------
// Edge MLP 1 (weight_conv hidden): m=relu([h_d||h_s]@W1+b1)@W2+b2 -> agg[dst]+=m
// B fragments register-resident (loop-invariant).
// ---------------------------------------------------------------------------
__global__ __launch_bounds__(128) void edge_mlp1_kernel(
    const unsigned short* __restrict__ hb, const int* __restrict__ src,
    const int* __restrict__ dst, const unsigned short* __restrict__ w1sw,
    const float* __restrict__ b1, const unsigned short* __restrict__ w2sw,
    const float* __restrict__ b2, float* __restrict__ agg, int E_) {
  __shared__ __align__(16) unsigned short sA[4][16 * 128];
  __shared__ __align__(16) unsigned short sM[4][16 * 64];
  __shared__ int sDst[4][16];

  const int tid = threadIdx.x, lane = tid & 31, w = tid >> 5;
  const int c0 = lane & 15, hi = (lane >> 4) ? 8 : 0;

  v16bf bW1[16], bW2[8];
  load_b_frags<128>(bW1, w1sw, lane);
  load_b_frags<64>(bW2, w2sw, lane);
  float b1v[4], b2v[4];
#pragma unroll
  for (int nt = 0; nt < 4; ++nt) {
    b1v[nt] = b1[nt * 16 + c0];
    b2v[nt] = b2[nt * 16 + c0];
  }

  const int numTiles = (E_ + 15) >> 4;
  for (int tb = blockIdx.x * 4; tb < numTiles; tb += gridDim.x * 4) {
    const int tile = tb + w;
    const bool active = tile < numTiles;
    if (active) stage_edge_tile(sA[w], sDst[w], hb, src, dst, tile, E_, lane);
    wait_async_lds();
    __syncthreads();

    if (active) {  // GEMM1: [16x128]x[128x64] + bias, relu -> sM (bf16)
      v8f acc[4];
      zero_acc4(acc);
      gemm_16x64_reg<128>(acc, sA[w], bW1, lane);
#pragma unroll
      for (int nt = 0; nt < 4; ++nt)
#pragma unroll
        for (int r = 0; r < 8; ++r) {
          float v = acc[nt][r] + b1v[nt];
          v = v > 0.f ? v : 0.f;
          sM[w][(hi + r) * 64 + nt * 16 + c0] = f32_to_bf16_rte(v);
        }
    }
    __syncthreads();

    if (active) {  // GEMM2: [16x64]x[64x64] + bias -> atomic scatter
      v8f acc[4];
      zero_acc4(acc);
      gemm_16x64_reg<64>(acc, sM[w], bW2, lane);
#pragma unroll
      for (int nt = 0; nt < 4; ++nt)
#pragma unroll
        for (int r = 0; r < 8; ++r) {
          const int d = sDst[w][hi + r];
          if (d >= 0)
            atomicAdd(&agg[d * 64 + nt * 16 + c0], acc[nt][r] + b2v[nt]);
        }
    }
    __syncthreads();
  }
}

// ---------------------------------------------------------------------------
// Edge MLP 2 (weight_conv scalar): relu([m_d||m_s]@W1+b1) . w2 + b2 -> agg[dst]
// ---------------------------------------------------------------------------
__global__ __launch_bounds__(128) void edge_mlp2_kernel(
    const unsigned short* __restrict__ mb, const int* __restrict__ src,
    const int* __restrict__ dst, const unsigned short* __restrict__ w1sw,
    const float* __restrict__ b1, const float* __restrict__ w2,
    const float* __restrict__ b2, float* __restrict__ agg, int E_) {
  __shared__ __align__(16) unsigned short sA[4][16 * 128];
  __shared__ float sMf[4][16 * 64];
  __shared__ float sV2[64];
  __shared__ int sDst[4][16];

  const int tid = threadIdx.x, lane = tid & 31, w = tid >> 5;
  const int c0 = lane & 15, hi = (lane >> 4) ? 8 : 0;

  v16bf bW1[16];
  load_b_frags<128>(bW1, w1sw, lane);
  float b1v[4];
#pragma unroll
  for (int nt = 0; nt < 4; ++nt) b1v[nt] = b1[nt * 16 + c0];
  if (tid < 64) sV2[tid] = w2[tid];
  const float bias2 = b2[0];
  __syncthreads();

  const int numTiles = (E_ + 15) >> 4;
  for (int tb = blockIdx.x * 4; tb < numTiles; tb += gridDim.x * 4) {
    const int tile = tb + w;
    const bool active = tile < numTiles;
    if (active) stage_edge_tile(sA[w], sDst[w], mb, src, dst, tile, E_, lane);
    wait_async_lds();
    __syncthreads();

    if (active) {
      v8f acc[4];
      zero_acc4(acc);
      gemm_16x64_reg<128>(acc, sA[w], bW1, lane);
#pragma unroll
      for (int nt = 0; nt < 4; ++nt)
#pragma unroll
        for (int r = 0; r < 8; ++r) {
          float v = acc[nt][r] + b1v[nt];
          sMf[w][(hi + r) * 64 + nt * 16 + c0] = v > 0.f ? v : 0.f;
        }
    }
    __syncthreads();

    if (active && lane < 16) {  // per-row dot with w2 vector
      const int d = sDst[w][lane];
      if (d >= 0) {
        float s = bias2;
#pragma unroll
        for (int k = 0; k < 64; ++k) s += sMf[w][lane * 64 + k] * sV2[k];
        atomicAdd(&agg[d], s);
      }
    }
    __syncthreads();
  }
}

// ---------------------------------------------------------------------------
// Node linear: out = (x @ W + b), optional relu; writes f32 + bf16 shadow.
// ---------------------------------------------------------------------------
__global__ __launch_bounds__(256) void node_linear_kernel(
    const float* __restrict__ xin, const unsigned short* __restrict__ wsw,
    const float* __restrict__ bias, float* __restrict__ out_f32,
    unsigned short* __restrict__ out_b16, int N_, int do_relu) {
  __shared__ __align__(16) unsigned short sA[8][16 * 64];

  const int tid = threadIdx.x, lane = tid & 31, w = tid >> 5;
  const int c0 = lane & 15, hi = (lane >> 4) ? 8 : 0;

  v16bf bW[8];
  load_b_frags<64>(bW, wsw, lane);
  float bv[4];
#pragma unroll
  for (int nt = 0; nt < 4; ++nt) bv[nt] = bias[nt * 16 + c0];

  const int numTiles = (N_ + 15) >> 4;
  for (int tb = blockIdx.x * 8; tb < numTiles; tb += gridDim.x * 8) {
    const int tile = tb + w;
    const bool active = tile < numTiles;
    if (active) stage_node_tile_f32(sA[w], xin, tile, N_, lane);
    __syncthreads();
    if (active) {
      v8f acc[4];
      zero_acc4(acc);
      gemm_16x64_reg<64>(acc, sA[w], bW, lane);
#pragma unroll
      for (int nt = 0; nt < 4; ++nt)
#pragma unroll
        for (int r = 0; r < 8; ++r) {
          const int row = tile * 16 + hi + r;
          if (row < N_) {
            const int col = nt * 16 + c0;
            float v = acc[nt][r] + bv[nt];
            if (do_relu) v = fmaxf(v, 0.f);
            out_f32[row * 64 + col] = v;
            if (out_b16) out_b16[row * 64 + col] = f32_to_bf16_rte(v);
          }
        }
    }
    __syncthreads();
  }
}

// ---------------------------------------------------------------------------
// Sparse conv transform: h = relu(agg @ Wrel + xm @ Wroot + brel + broot)
// ---------------------------------------------------------------------------
__global__ __launch_bounds__(256) void sparse_conv_kernel(
    const float* __restrict__ agg, const unsigned short* __restrict__ xmb,
    const unsigned short* __restrict__ wrelsw, const unsigned short* __restrict__ wrootsw,
    const float* __restrict__ brel, const float* __restrict__ broot,
    float* __restrict__ h_f32, unsigned short* __restrict__ h_b16, int N_) {
  __shared__ __align__(16) unsigned short sA1[8][16 * 64];
  __shared__ __align__(16) unsigned short sA2[8][16 * 64];

  const int tid = threadIdx.x, lane = tid & 31, w = tid >> 5;
  const int c0 = lane & 15, hi = (lane >> 4) ? 8 : 0;

  v16bf bW1[8], bW2[8];
  load_b_frags<64>(bW1, wrelsw, lane);
  load_b_frags<64>(bW2, wrootsw, lane);
  float bv[4];
#pragma unroll
  for (int nt = 0; nt < 4; ++nt)
    bv[nt] = brel[nt * 16 + c0] + broot[nt * 16 + c0];

  const int numTiles = (N_ + 15) >> 4;
  for (int tb = blockIdx.x * 8; tb < numTiles; tb += gridDim.x * 8) {
    const int tile = tb + w;
    const bool active = tile < numTiles;
    if (active) {
      stage_node_tile_f32(sA1[w], agg, tile, N_, lane);
      stage_node_tile_bf16(sA2[w], xmb, tile, N_, lane);
    }
    wait_async_lds();
    __syncthreads();
    if (active) {
      v8f acc[4];
      zero_acc4(acc);
      gemm_16x64_reg<64>(acc, sA1[w], bW1, lane);   // agg @ Wrel
      gemm_16x64_reg<64>(acc, sA2[w], bW2, lane);   // + xm @ Wroot
#pragma unroll
      for (int nt = 0; nt < 4; ++nt)
#pragma unroll
        for (int r = 0; r < 8; ++r) {
          const int row = tile * 16 + hi + r;
          if (row < N_) {
            const int col = nt * 16 + c0;
            float v = fmaxf(acc[nt][r] + bv[nt], 0.f);
            h_f32[row * 64 + col] = v;
            h_b16[row * 64 + col] = f32_to_bf16_rte(v);
          }
        }
    }
    __syncthreads();
  }
}

// ---------------------------------------------------------------------------
// Elementwise / scatter helpers
// ---------------------------------------------------------------------------
__global__ void zero_f32_kernel(float* p, int n) {
  int i = blockIdx.x * blockDim.x + threadIdx.x;
  if (i < n) p[i] = 0.f;
}

// f32 -> bf16 conversion fused with fragment-native swizzle (global output).
// Output per matrix: frag(kb,nt): 32 lanes x 16 contiguous values.
template <int KD>
__global__ void cvt_swizzle_kernel(const float* __restrict__ in,
                                   unsigned short* __restrict__ out, int nmat) {
  const int total = KD * 64;
  int idx = blockIdx.x * blockDim.x + threadIdx.x;
  if (idx < nmat * total) {
    const int m = idx / total;
    const int i = idx % total;
    const int frag = i >> 9, lane = (i >> 4) & 31, j = i & 15;
    const int kb = frag >> 2, nt = frag & 3;
    const int ko = (lane >> 4) ? 8 : 0;
    const int col = nt * 16 + (lane & 15);
    out[idx] = f32_to_bf16_rte(in[m * total + (kb * 32 + ko + (j & 7) + ((j & 8) << 1)) * 64 + col]);
  }
}

__global__ void degree_kernel(const int* __restrict__ dst, float* __restrict__ cnt, int E_) {
  int e = blockIdx.x * blockDim.x + threadIdx.x;
  if (e < E_) atomicAdd(&cnt[dst[e]], 1.0f);
}

// mask1 = sigmoid(agg / max(cnt,1)) -> bf16
__global__ void mask1_kernel(const float* __restrict__ agg, const float* __restrict__ cnt,
                             unsigned short* __restrict__ mb, int total) {
  int i = blockIdx.x * blockDim.x + threadIdx.x;
  if (i < total) {
    float c = fmaxf(cnt[i >> 6], 1.0f);
    float v = agg[i] / c;
    float s = 1.0f / (1.0f + __expf(-v));
    mb[i] = f32_to_bf16_rte(s);
  }
}

// xm = h * sigmoid(agg2 / max(cnt,1)) -> f32 + bf16
__global__ void mask_apply_kernel(const float* __restrict__ h, const float* __restrict__ agg2,
                                  const float* __restrict__ cnt, float* __restrict__ xmf,
                                  unsigned short* __restrict__ xmb, int total) {
  int i = blockIdx.x * blockDim.x + threadIdx.x;
  if (i < total) {
    int n = i >> 6;
    float c = fmaxf(cnt[n], 1.0f);
    float mv = 1.0f / (1.0f + __expf(-(agg2[n] / c)));
    float v = h[i] * mv;
    xmf[i] = v;
    xmb[i] = f32_to_bf16_rte(v);
  }
}

// agg3[dst] += xm[src], 4 columns per thread
__global__ void scatter_kernel(const float* __restrict__ xmf, const int* __restrict__ src,
                               const int* __restrict__ dst, float* __restrict__ agg3, int E_) {
  int i = blockIdx.x * blockDim.x + threadIdx.x;
  int e = i >> 4;
  if (e < E_) {
    int c4 = (i & 15) * 4;
    int s = src[e], d = dst[e];
    if (e + 1 < E_) __builtin_prefetch(&xmf[src[e + 1] * 64 + c4], 0, 1);
    float4 v = *(const float4*)(xmf + s * 64 + c4);
    float* p = agg3 + d * 64 + c4;
    atomicAdd(p + 0, v.x);
    atomicAdd(p + 1, v.y);
    atomicAdd(p + 2, v.z);
    atomicAdd(p + 3, v.w);
  }
}

// pooled[batch[n]] += h[n]
__global__ void pool_kernel(const float* __restrict__ h, const int* __restrict__ batch,
                            float* __restrict__ pooled, int N_) {
  int i = blockIdx.x * blockDim.x + threadIdx.x;
  int n = i >> 4;
  if (n < N_) {
    int c4 = (i & 15) * 4;
    int g = batch[n];
    float4 v = *(const float4*)(h + n * 64 + c4);
    float* p = pooled + g * 64 + c4;
    atomicAdd(p + 0, v.x);
    atomicAdd(p + 1, v.y);
    atomicAdd(p + 2, v.z);
    atomicAdd(p + 3, v.w);
  }
}

// out = log_softmax(relu(pooled @ W1 + b1) @ W2 + b2); one block per graph
__global__ __launch_bounds__(64) void head_kernel(
    const float* __restrict__ pooled, const float* __restrict__ W1,
    const float* __restrict__ b1, const float* __restrict__ W2,
    const float* __restrict__ b2, float* __restrict__ out, int C_) {
  __shared__ float sp[64], hid[64], o[64], red[2];
  const int g = blockIdx.x, t = threadIdx.x;
  sp[t] = pooled[g * 64 + t];
  __syncthreads();
  float s = b1[t];
  for (int k = 0; k < 64; ++k) s += sp[k] * W1[k * 64 + t];
  hid[t] = fmaxf(s, 0.f);
  __syncthreads();
  if (t < C_) {
    float s2 = b2[t];
    for (int k = 0; k < 64; ++k) s2 += hid[k] * W2[k * C_ + t];
    o[t] = s2;
  }
  __syncthreads();
  if (t == 0) {
    float mx = o[0];
    for (int c = 1; c < C_; ++c) mx = fmaxf(mx, o[c]);
    float se = 0.f;
    for (int c = 0; c < C_; ++c) se += __expf(o[c] - mx);
    red[0] = mx; red[1] = __logf(se);
  }
  __syncthreads();
  if (t < C_) out[g * C_ + t] = o[t] - red[0] - red[1];
}

// ---------------------------------------------------------------------------
// Host launcher
// ---------------------------------------------------------------------------
static inline size_t align256(size_t x) { return (x + 255) & ~(size_t)255; }

extern "C" void kernel_launch(void* const* d_in, const int* in_sizes, int n_in,
                              void* d_out, int out_size, void* d_ws, size_t ws_size,
                              hipStream_t stream) {
  const float* x       = (const float*)d_in[0];
  const int*  eidx     = (const int*)d_in[1];
  const int*  batch    = (const int*)d_in[2];
  const float* W0      = (const float*)d_in[3];
  const float* b0      = (const float*)d_in[4];
  const float* ma1_w1  = (const float*)d_in[5];
  const float* ma1_b1  = (const float*)d_in[6];
  const float* ma1_w2  = (const float*)d_in[7];
  const float* ma1_b2  = (const float*)d_in[8];
  const float* ma2_w1  = (const float*)d_in[9];
  const float* ma2_b1  = (const float*)d_in[10];
  const float* ma2_w2  = (const float*)d_in[11];
  const float* ma2_b2  = (const float*)d_in[12];
  const float* wrel    = (const float*)d_in[13];
  const float* brel    = (const float*)d_in[14];
  const float* wroot   = (const float*)d_in[15];
  const float* broot   = (const float*)d_in[16];
  const float* W1      = (const float*)d_in[17];
  const float* b1      = (const float*)d_in[18];
  const float* W2      = (const float*)d_in[19];
  const float* b2      = (const float*)d_in[20];

  const int N = in_sizes[2];
  const int E = in_sizes[1] / 2;
  const int H = 64;                       // fixed by design (WMMA tiling)
  const int C = in_sizes[20];
  const int Lnum = in_sizes[6] / H;
  const int G = out_size / C;
  const int* src = eidx;
  const int* dst = eidx + E;

  // ---- workspace carve-out ----
  char* base = (char*)d_ws;
  size_t off = 0;
  auto alloc = [&](size_t bytes) { char* p = base + off; off += align256(bytes); return p; };
  float* h_f            = (float*)alloc((size_t)N * H * 4);
  unsigned short* h_b   = (unsigned short*)alloc((size_t)N * H * 2);
  float* agg1           = (float*)alloc((size_t)N * H * 4);
  float* cnt            = (float*)alloc((size_t)N * 4);
  unsigned short* m1b   = (unsigned short*)alloc((size_t)N * H * 2);
  float* agg2           = (float*)alloc((size_t)N * 4);
  float* xmf            = (float*)alloc((size_t)N * H * 4);
  unsigned short* xmb   = (unsigned short*)alloc((size_t)N * H * 2);
  float* agg3           = (float*)alloc((size_t)N * H * 4);
  float* pooled         = (float*)alloc((size_t)G * H * 4);
  unsigned short* w0b   = (unsigned short*)alloc((size_t)H * H * 2);
  unsigned short* m1w1b = (unsigned short*)alloc((size_t)Lnum * 2 * H * H * 2);
  unsigned short* m1w2b = (unsigned short*)alloc((size_t)Lnum * H * H * 2);
  unsigned short* m2w1b = (unsigned short*)alloc((size_t)Lnum * 2 * H * H * 2);
  unsigned short* wrelb = (unsigned short*)alloc((size_t)Lnum * H * H * 2);
  unsigned short* wrootb= (unsigned short*)alloc((size_t)Lnum * H * H * 2);
  (void)ws_size; (void)n_in;

  auto gridFor = [](int n, int blk) { return (n + blk - 1) / blk; };

  // ---- weight conversion to bf16 + fragment-native swizzle ----
  cvt_swizzle_kernel<64><<<gridFor(H * H, 256), 256, 0, stream>>>(W0, w0b, 1);
  cvt_swizzle_kernel<128><<<gridFor(Lnum * 2 * H * H, 256), 256, 0, stream>>>(ma1_w1, m1w1b, Lnum);
  cvt_swizzle_kernel<64><<<gridFor(Lnum * H * H, 256), 256, 0, stream>>>(ma1_w2, m1w2b, Lnum);
  cvt_swizzle_kernel<128><<<gridFor(Lnum * 2 * H * H, 256), 256, 0, stream>>>(ma2_w1, m2w1b, Lnum);
  cvt_swizzle_kernel<64><<<gridFor(Lnum * H * H, 256), 256, 0, stream>>>(wrel, wrelb, Lnum);
  cvt_swizzle_kernel<64><<<gridFor(Lnum * H * H, 256), 256, 0, stream>>>(wroot, wrootb, Lnum);

  // ---- in-degree (layer invariant) ----
  zero_f32_kernel<<<gridFor(N, 256), 256, 0, stream>>>(cnt, N);
  degree_kernel<<<gridFor(E, 256), 256, 0, stream>>>(dst, cnt, E);

  // ---- h = x @ W0 + b0 ----
  const int nodeTiles = gridFor(N, 16);
  const int nodeBlocks = gridFor(nodeTiles, 8);
  node_linear_kernel<<<nodeBlocks, 256, 0, stream>>>(x, w0b, b0, h_f, h_b, N, /*relu=*/0);

  const int edgeTiles = gridFor(E, 16);
  const int edgeBlocks = gridFor(edgeTiles, 4);

  for (int i = 0; i < Lnum; ++i) {
    // weight_conv 1 -> mask_1ho (bf16)
    zero_f32_kernel<<<gridFor(N * H, 256), 256, 0, stream>>>(agg1, N * H);
    edge_mlp1_kernel<<<edgeBlocks, 128, 0, stream>>>(
        h_b, src, dst, m1w1b + (size_t)i * 2 * H * H, ma1_b1 + (size_t)i * H,
        m1w2b + (size_t)i * H * H, ma1_b2 + (size_t)i * H, agg1, E);
    mask1_kernel<<<gridFor(N * H, 256), 256, 0, stream>>>(agg1, cnt, m1b, N * H);

    // weight_conv 2 -> mask_val (scalar per node)
    zero_f32_kernel<<<gridFor(N, 256), 256, 0, stream>>>(agg2, N);
    edge_mlp2_kernel<<<edgeBlocks, 128, 0, stream>>>(
        m1b, src, dst, m2w1b + (size_t)i * 2 * H * H, ma2_b1 + (size_t)i * H,
        ma2_w2 + (size_t)i * H, ma2_b2 + i, agg2, E);

    // xm = h * sigmoid(mask_val); neighbor sum; output transform
    mask_apply_kernel<<<gridFor(N * H, 256), 256, 0, stream>>>(h_f, agg2, cnt, xmf, xmb, N * H);
    zero_f32_kernel<<<gridFor(N * H, 256), 256, 0, stream>>>(agg3, N * H);
    scatter_kernel<<<gridFor(E * 16, 256), 256, 0, stream>>>(xmf, src, dst, agg3, E);
    sparse_conv_kernel<<<nodeBlocks, 256, 0, stream>>>(
        agg3, xmb, wrelb + (size_t)i * H * H, wrootb + (size_t)i * H * H,
        brel + (size_t)i * H, broot + (size_t)i * H, h_f, h_b, N);
  }

  // ---- global add pool + classifier head ----
  zero_f32_kernel<<<gridFor(G * H, 256), 256, 0, stream>>>(pooled, G * H);
  pool_kernel<<<gridFor(N * 16, 256), 256, 0, stream>>>(h_f, batch, pooled, N);
  head_kernel<<<G, 64, 0, stream>>>(pooled, W1, b1, W2, b2, (float*)d_out, C);
}
